// FaberConv_28664611733983
// MI455X (gfx1250) — compile-verified, matrix operand order checked by reading, not run on hardware
//
#include <hip/hip_runtime.h>
#include <hip/hip_bf16.h>

// FaberConv on MI455X (gfx1250, wave32).
//   N=100000 nodes, E=1600000 edges, D=64, K_PLUS=4, EXPONENT=0.5, ALPHA=0.5,
//   WEIGHT_PENALTY='exp', ZERO_ORDER=True.
// SpMM hops are bandwidth/atomic bound; dense 64x64 projections run on
// v_wmma_f32_16x16x32_bf16 with f32 accumulate, weights staged in LDS as bf16
// (hop coefficients -- all powers of two -- folded into the staged weights).
// f32->bf16x2 pack: 2x(and+add3) RNE + one v_perm_b32 combine.

#define NN 100000
#define EE 1600000
#define DD 64
#define KPLUS 4
#define ALPHA_C 0.5f

typedef __attribute__((ext_vector_type(16))) __bf16 v16bf;
typedef __attribute__((ext_vector_type(8)))  float  v8f;

union BfVec { v16bf v; unsigned int u[8]; };

__device__ __forceinline__ unsigned int pack_bf16x2(float a, float b) {
  // round-to-nearest-even f32 -> bf16 in the high half of each word,
  // then one v_perm_b32 gathers {ub[3],ub[2],ua[3],ua[2]} (elem0 low).
  unsigned int ua = __float_as_uint(a), ub = __float_as_uint(b);
  ua += 0x7fffu + ((ua >> 16) & 1u);
  ub += 0x7fffu + ((ub >> 16) & 1u);
  return __builtin_amdgcn_perm(ub, ua, 0x07060302u);
}

// ---------------- degree + directional norm ----------------

__global__ void deg_kernel(const int* __restrict__ row, const int* __restrict__ col,
                           float* __restrict__ od, float* __restrict__ idg) {
  int e = blockIdx.x * blockDim.x + threadIdx.x;
  if (e < EE) {
    atomicAdd(&od[row[e]], 1.0f);
    atomicAdd(&idg[col[e]], 1.0f);
  }
}

__global__ void vals_kernel(const int* __restrict__ row, const int* __restrict__ col,
                            const float* __restrict__ od, const float* __restrict__ idg,
                            float* __restrict__ vals, float* __restrict__ vals_t) {
  int e = blockIdx.x * blockDim.x + threadIdx.x;
  if (e < EE) {
    float o = od[row[e]];   // out-degree of source
    float i = idg[col[e]];  // in-degree of dest
    // EXPONENT = 0.5 and 1-EXPONENT = 0.5 -> both are rsqrt
    float on = (o > 0.0f) ? rsqrtf(o) : 0.0f;
    float in = (i > 0.0f) ? rsqrtf(i) : 0.0f;
    vals[e] = on * in;
    // _dir_norm_vals(col,row): out_deg'=in_deg, in_deg'=out_deg -> same product
    vals_t[e] = in * on;
  }
}

// ---------------- SpMM: out[dst] += vals * h[src] (16 threads / edge) ----------------

__global__ void spmm_kernel(const float* __restrict__ vals, const int* __restrict__ src,
                            const int* __restrict__ dst, const float* __restrict__ h,
                            float* __restrict__ out) {
  unsigned int tid = blockIdx.x * blockDim.x + threadIdx.x;
  unsigned int e = tid >> 4;
  if (e >= (unsigned int)EE) return;
  int f = (int)(tid & 15u) << 2;
  float v = vals[e];
  const float4 hv = *(const float4*)(h + (size_t)src[e] * DD + f);
  float* o = out + (size_t)dst[e] * DD + f;
  atomicAdd(o + 0, v * hv.x);
  atomicAdd(o + 1, v * hv.y);
  atomicAdd(o + 2, v * hv.z);
  atomicAdd(o + 3, v * hv.w);
}

// ---------------- fused multi-term WMMA GEMM: out (+)= sum_j c_j*(H_j @ W_j^T + b_j) ----------------

struct GemmArgs {
  const float* H[4];
  const float* W[4];  // [64][64], row = out feature, col = in feature
  const float* b[4];  // [64]
  float c[4];         // powers of two -> exact when folded into bf16 weights
  int nterms;
  int init;  // 1: overwrite out, 0: accumulate into out
};

__global__ void __launch_bounds__(256)
gemm_wmma_kernel(GemmArgs args, float* __restrict__ out) {
  // bf16 weights (pre-scaled by c_j) for up to 4 terms:
  // [term][n(64)][kpair(32)] packed bf16 pairs
  __shared__ unsigned int wlds[4 * 2048];
  __shared__ float blds[64];  // combined bias for this call

  const int t = threadIdx.x;

  for (int j = 0; j < args.nterms; ++j) {
    const float* Wj = args.W[j];
    const float cj = args.c[j];
#pragma unroll
    for (int it = 0; it < 8; ++it) {
      int idx = t * 8 + it;  // 0..2047 == n*32 + kpair
      float2 wv = *(const float2*)(Wj + idx * 2);
      wlds[j * 2048 + idx] = pack_bf16x2(cj * wv.x, cj * wv.y);
    }
  }
  if (t < 64) {
    float bs = 0.0f;
    for (int j = 0; j < args.nterms; ++j) bs += args.c[j] * args.b[j][t];
    blds[t] = bs;
  }
  __syncthreads();

  const int wave = t >> 5;
  const int lane = t & 31;
  const int tile = blockIdx.x * 8 + wave;   // 16-row node tile
  if (tile >= NN / 16) return;              // wave-uniform: EXEC stays all-1s
  const int nl = lane & 15;
  const int kh = lane >> 4;                 // lane-half selects K/M group
  const int mrow = tile * 16;

  // C/D layout: VGPR r, lane -> (m = r + 8*kh, n = nl)
  v8f acc[4];
#pragma unroll
  for (int ct = 0; ct < 4; ++ct) {
    float bias = blds[ct * 16 + nl];
#pragma unroll
    for (int r = 0; r < 8; ++r) {
      float c0 = bias;
      if (!args.init)
        c0 += out[(size_t)(mrow + r + 8 * kh) * DD + ct * 16 + nl];
      acc[ct][r] = c0;
    }
  }

  for (int j = 0; j < args.nterms; ++j) {
    const float* hrow = args.H[j] + (size_t)(mrow + nl) * DD;  // A row m = nl
    BfVec afr[2];
    // A 16x32 bf16 layout: lane(m=nl, half=kh); VGPR v -> k = ks*32 + (v>=4?16:0) + kh*8 + (v&3)*2
#pragma unroll
    for (int ks = 0; ks < 2; ++ks) {
#pragma unroll
      for (int v = 0; v < 8; ++v) {
        int k = ks * 32 + ((v >= 4) ? 16 : 0) + kh * 8 + (v & 3) * 2;
        float2 hv = *(const float2*)(hrow + k);
        afr[ks].u[v] = pack_bf16x2(hv.x, hv.y);
      }
    }
#pragma unroll
    for (int ct = 0; ct < 4; ++ct) {
#pragma unroll
      for (int ks = 0; ks < 2; ++ks) {
        BfVec bfr;
        // B 32x16: B[k][n] = c_j*W[n][k]; mirror A striping with n in place of m
#pragma unroll
        for (int v = 0; v < 8; ++v) {
          int k = ks * 32 + ((v >= 4) ? 16 : 0) + kh * 8 + (v & 3) * 2;
          bfr.u[v] = wlds[j * 2048 + (ct * 16 + nl) * 32 + (k >> 1)];
        }
        acc[ct] = __builtin_amdgcn_wmma_f32_16x16x32_bf16(
            false, afr[ks].v, false, bfr.v, (short)0, acc[ct], false, false);
      }
    }
  }

#pragma unroll
  for (int ct = 0; ct < 4; ++ct) {
#pragma unroll
    for (int r = 0; r < 8; ++r) {
      out[(size_t)(mrow + r + 8 * kh) * DD + ct * 16 + nl] = acc[ct][r];
    }
  }
}

// ---------------- driver ----------------

extern "C" void kernel_launch(void* const* d_in, const int* in_sizes, int n_in,
                              void* d_out, int out_size, void* d_ws, size_t ws_size,
                              hipStream_t stream) {
  (void)in_sizes; (void)n_in; (void)out_size; (void)ws_size;

  const float* x     = (const float*)d_in[0];
  const int*   ei    = (const int*)d_in[1];
  const int*   row   = ei;
  const int*   col   = ei + EE;
  const float* Ws_sd = (const float*)d_in[2];
  const float* bs_sd = (const float*)d_in[3];
  const float* Ws_ds = (const float*)d_in[4];
  const float* bs_ds = (const float*)d_in[5];
  const float* W0_sd = (const float*)d_in[6];
  const float* b0_sd = (const float*)d_in[7];
  const float* W0_ds = (const float*)d_in[8];
  const float* b0_ds = (const float*)d_in[9];
  float* out = (float*)d_out;

  float* ws     = (float*)d_ws;
  float* od     = ws;                       // [N]
  float* idg    = od + NN;                  // [N]
  float* vals   = idg + NN;                 // [E]
  float* vals_t = vals + EE;                // [E]
  float* yA     = vals_t + EE;              // [N*64]
  float* yB     = yA + (size_t)NN * DD;     // [N*64]
  float* yt     = yB + (size_t)NN * DD;     // [N*64]
  const size_t featB = (size_t)NN * DD * sizeof(float);

  const int eblk = (EE + 255) / 256;
  const int sblk = (EE * 16 + 255) / 256;   // 16 threads / edge
  const int gblk = (NN / 16 + 7) / 8;       // 8 waves/block, 1 tile/wave

  // degrees + norm values
  hipMemsetAsync(od, 0, 2 * NN * sizeof(float), stream);
  deg_kernel<<<eblk, 256, 0, stream>>>(row, col, od, idg);
  vals_kernel<<<eblk, 256, 0, stream>>>(row, col, od, idg, vals, vals_t);

  // hop 1: y1 = A x ; yt1 = At x
  hipMemsetAsync(yA, 0, featB, stream);
  spmm_kernel<<<sblk, 256, 0, stream>>>(vals, col, row, x, yA);     // out[row]+=v*h[col]
  hipMemsetAsync(yt, 0, featB, stream);
  spmm_kernel<<<sblk, 256, 0, stream>>>(vals_t, row, col, x, yt);   // out[col]+=v*h[row]

  // out = A*(x@W0_sd^T+b) + (1-A)*(x@W0_ds^T+b) + A*(y1@Ws_sd0^T+b) + (1-A)*(yt1@Ws_ds0^T+b)
  GemmArgs g0{};
  g0.H[0] = x;  g0.W[0] = W0_sd; g0.b[0] = b0_sd; g0.c[0] = ALPHA_C;
  g0.H[1] = x;  g0.W[1] = W0_ds; g0.b[1] = b0_ds; g0.c[1] = 1.0f - ALPHA_C;
  g0.H[2] = yA; g0.W[2] = Ws_sd; g0.b[2] = bs_sd; g0.c[2] = ALPHA_C;
  g0.H[3] = yt; g0.W[3] = Ws_ds; g0.b[3] = bs_ds; g0.c[3] = 1.0f - ALPHA_C;
  g0.nterms = 4; g0.init = 1;
  gemm_wmma_kernel<<<gblk, 256, 0, stream>>>(g0, out);

  // hops 2..4: y_{i+1} = A y_i ; yt_{i+1} = At y_{i+1} (faithful: uses updated y)
  float* ycur = yA;
  float* yoth = yB;
  for (int i = 1; i < KPLUS; ++i) {
    float* ynew = yoth;
    hipMemsetAsync(ynew, 0, featB, stream);
    spmm_kernel<<<sblk, 256, 0, stream>>>(vals, col, row, ycur, ynew);
    hipMemsetAsync(yt, 0, featB, stream);
    spmm_kernel<<<sblk, 256, 0, stream>>>(vals_t, row, col, ynew, yt);

    float sc = 1.0f / (float)(1 << i);  // WEIGHT_PENALTY='exp'
    GemmArgs g{};
    g.H[0] = ynew; g.W[0] = Ws_sd + (size_t)i * DD * DD; g.b[0] = bs_sd + i * DD; g.c[0] = ALPHA_C * sc;
    g.H[1] = yt;   g.W[1] = Ws_ds + (size_t)i * DD * DD; g.b[1] = bs_ds + i * DD; g.c[1] = (1.0f - ALPHA_C) * sc;
    g.H[2] = x;    g.W[2] = W0_sd; g.b[2] = b0_sd; g.c[2] = 0.0f;  // unused (nterms=2)
    g.H[3] = x;    g.W[3] = W0_ds; g.b[3] = b0_ds; g.c[3] = 0.0f;
    g.nterms = 2; g.init = 0;
    gemm_wmma_kernel<<<gblk, 256, 0, stream>>>(g, out);

    yoth = ycur;
    ycur = ynew;
  }
}
